// CustomLSTM_39264591020246
// MI455X (gfx1250) — matrix operand order repaired and checked by reading, not compiled, not run on hardware
//
#include <hip/hip_runtime.h>
#include <hip/hip_bf16.h>
#include <stdint.h>

// ---------------------------------------------------------------------------
// CDNA5 (gfx1250) LSTM forward.
//  Phase 1: xg[t,b,g] = x @ w_ih^T + (b_ih+b_hh)    (one big bf16 WMMA GEMM)
//  Phase 2: 512 sequential step kernels, each: gates = xg_t + h @ w_hh^T
//           (bf16 WMMA, fp32 accum, h tile staged in LDS), fused activations.
// ---------------------------------------------------------------------------

typedef __bf16 bf16_t;
typedef __bf16 bf16x4 __attribute__((ext_vector_type(4)));
typedef __bf16 bf16x8 __attribute__((ext_vector_type(8)));
typedef __bf16 v16bf  __attribute__((ext_vector_type(16)));
typedef float  v8f    __attribute__((ext_vector_type(8)));

#define T_STEPS 512
#define BATCH   64
#define ISZ     1024
#define HSZ     1024
#define GSZ     (4 * HSZ) /* 4096 */
#define LDS_PITCH 1032    /* 1024 + 8 bf16 pad: 16-row b128 reads hit all 64 banks */

__device__ __forceinline__ float sigmoidf_(float x) {
  return 1.0f / (1.0f + __expf(-x));
}

// ---------------- fp32 -> bf16 conversion (4-wide) -------------------------
__global__ void cvt4_f32_bf16(const float4* __restrict__ s,
                              bf16x4* __restrict__ d, int n4) {
  int i = blockIdx.x * blockDim.x + threadIdx.x;
  if (i < n4) {
    float4 v = s[i];
    bf16x4 o;
    o[0] = (bf16_t)v.x; o[1] = (bf16_t)v.y;
    o[2] = (bf16_t)v.z; o[3] = (bf16_t)v.w;
    d[i] = o;
  }
}

__global__ void init_state(const float* __restrict__ h0,
                           const float* __restrict__ c0,
                           bf16_t* __restrict__ hb,
                           float* __restrict__ cs, int n) {
  int i = blockIdx.x * blockDim.x + threadIdx.x;
  if (i < n) { hb[i] = (bf16_t)h0[i]; cs[i] = c0[i]; }
}

// -------- fragment loaders (ISA 7.12.2 16-bit A 16x32 / B 32x16 layout) ----
// A: lane(m = L&15, hi = L>>4): j=0..7 -> k=8*hi+j ; j=8..15 -> k=16+8*hi+(j-8)
__device__ __forceinline__ v16bf load_a_frag(const bf16_t* base) {
  bf16x8 a0 = *(const bf16x8*)(base);
  bf16x8 a1 = *(const bf16x8*)(base + 16);
  v16bf f;
#pragma unroll
  for (int j = 0; j < 8; ++j) { f[j] = a0[j]; f[j + 8] = a1[j]; }
  return f;
}
// B: lane(n = L&15, hi = L>>4): j=0..7 -> k=8*hi+j ; j=8..15 -> k=16+8*hi+(j-8)
__device__ __forceinline__ v16bf load_b_frag(const bf16_t* __restrict__ base) {
  bf16x8 b0 = *(const bf16x8*)(base);
  bf16x8 b1 = *(const bf16x8*)(base + 16);
  v16bf f;
#pragma unroll
  for (int j = 0; j < 8; ++j) { f[j] = b0[j]; f[j + 8] = b1[j]; }
  return f;
}

// ---------------- Phase 1: xg = x @ w_ih^T + bias --------------------------
// A: x_bf  [M=T*B=32768, K=1024]   B: w_ih_bf [N=4096, K=1024] (row n = col n)
// One 16x16 tile per wave; 8 waves / block; 65536 blocks.
__global__ __launch_bounds__(256) void gemm_xg(
    const bf16_t* __restrict__ A, const bf16_t* __restrict__ Bw,
    const float* __restrict__ b_ih, const float* __restrict__ b_hh,
    float* __restrict__ xg) {
  const int K = ISZ;
  int wid  = blockIdx.x * (blockDim.x >> 5) + (threadIdx.x >> 5);
  int mt   = wid & 2047;         // 32768/16 = 2048 M tiles
  int nt   = wid >> 11;          // 4096/16  = 256  N tiles
  int bm   = mt << 4, bn = nt << 4;
  int lane = threadIdx.x & 31;
  int lm   = lane & 15, lhi = lane >> 4;

  const bf16_t* arow = A  + (size_t)(bm + lm) * K + 8  * lhi;
  const bf16_t* brow = Bw + (size_t)(bn + lm) * K + 8 * lhi;

  v8f acc = {};
#pragma unroll 4
  for (int k0 = 0; k0 < K; k0 += 32) {
    __builtin_prefetch((const void*)(arow + k0 + 128), 0, 3);  // WGP-scope
    v16bf af = load_a_frag(arow + k0);
    v16bf bf = load_b_frag(brow + k0);
    acc = __builtin_amdgcn_wmma_f32_16x16x32_bf16(
        false, af, false, bf, (short)0, acc, false, false);
  }

  float bias = b_ih[bn + lm] + b_hh[bn + lm];
#pragma unroll
  for (int r = 0; r < 8; ++r) {
    int row = bm + r + 8 * lhi;                // C/D layout: lanes16-31 -> M+8
    xg[(size_t)row * GSZ + bn + lm] = acc[r] + bias;
  }
}

// ---------------- Phase 2: one recurrent step ------------------------------
// 8-wave workgroups sharing one 16-row batch tile of h, staged once in LDS.
// grid = 32 blocks: blockIdx = mt(0..3) + 4*hgroup(0..7); wave w -> hidden
// tile hb = hgroup*128 + w*16. Each wave accumulates all 4 gate tiles
// (i/f/g/o) sharing the same LDS-resident A fragment.
__global__ __launch_bounds__(256) void lstm_step(
    const float*  __restrict__ xg_t,   // [64, 4096]
    const bf16_t* __restrict__ h_in,   // [64, 1024] bf16 (prev h)
    const bf16_t* __restrict__ Whh,    // [4096, 1024] bf16
    bf16_t* __restrict__ h_out,        // [64, 1024] bf16 (next h)
    float*  __restrict__ c_state,      // [64, 1024] fp32 (in/out)
    float*  __restrict__ out_h,        // outputs[t]
    float*  __restrict__ out_i, float* __restrict__ out_f,
    float*  __restrict__ out_g, float* __restrict__ out_o,
    float*  __restrict__ hT,  float* __restrict__ cT) {   // non-null on last t
  const int K = HSZ;
  __shared__ bf16_t hsh[16][LDS_PITCH];        // ~33 KB of the 320 KB LDS

  int bm   = (blockIdx.x & 3) << 4;            // batch tile base (shared)
  int hgrp = (blockIdx.x >> 2) << 7;           // 128-wide hidden group
  int wv   = threadIdx.x >> 5;
  int hb   = hgrp + (wv << 4);                 // this wave's hidden tile
  int lane = threadIdx.x & 31;
  int lm   = lane & 15, lhi = lane >> 4;

  // ---- cooperative stage: h rows [bm, bm+16) x 1024 -> LDS (b128 chunks) --
  for (int idx = threadIdx.x; idx < 16 * 128; idx += 256) {
    int row = idx >> 7;                        // 128 16B-chunks per row
    int ch  = idx & 127;
    *(bf16x8*)&hsh[row][ch * 8] =
        *(const bf16x8*)(h_in + (size_t)(bm + row) * K + ch * 8);
  }
  __syncthreads();

  // warm the cache for the post-GEMM xg tile while the K loop runs
  {
    size_t gbase = (size_t)(bm + lm) * GSZ + hb;
#pragma unroll
    for (int g = 0; g < 4; ++g)
      __builtin_prefetch((const void*)(xg_t + gbase + g * HSZ), 0, 3);
  }

  const bf16_t* arow = &hsh[lm][8 * lhi];      // A fragments from LDS
  const bf16_t* brow[4];
#pragma unroll
  for (int g = 0; g < 4; ++g)
    brow[g] = Whh + (size_t)(g * HSZ + hb + lm) * K + 8 * lhi;

  v8f acc[4] = {};
  for (int k0 = 0; k0 < K; k0 += 32) {
    v16bf af = load_a_frag(arow + k0);
#pragma unroll
    for (int g = 0; g < 4; ++g) {
      v16bf bf = load_b_frag(brow[g] + k0);
      acc[g] = __builtin_amdgcn_wmma_f32_16x16x32_bf16(
          false, af, false, bf, (short)0, acc[g], false, false);
    }
  }

#pragma unroll
  for (int r = 0; r < 8; ++r) {
    int row = bm + r + 8 * lhi;
    int col = hb + lm;
    size_t gbase = (size_t)row * GSZ + col;
    float iv = acc[0][r] + xg_t[gbase];
    float fv = acc[1][r] + xg_t[gbase + HSZ];
    float gv = acc[2][r] + xg_t[gbase + 2 * HSZ];
    float ov = acc[3][r] + xg_t[gbase + 3 * HSZ];

    float ia = sigmoidf_(iv);
    float fa = sigmoidf_(fv);
    float ga = tanhf(gv);
    float oa = sigmoidf_(ov);

    size_t idx = (size_t)row * HSZ + col;
    float cn = fa * c_state[idx] + ia * ga;
    float hn = oa * tanhf(cn);

    c_state[idx] = cn;
    h_out[idx]   = (bf16_t)hn;
    out_h[idx]   = hn;
    out_i[idx]   = ia;
    out_f[idx]   = fa;
    out_g[idx]   = ga;
    out_o[idx]   = oa;
    if (hT) { hT[idx] = hn; cT[idx] = cn; }
  }
}

// ---------------------------------------------------------------------------
extern "C" void kernel_launch(void* const* d_in, const int* in_sizes, int n_in,
                              void* d_out, int out_size, void* d_ws,
                              size_t ws_size, hipStream_t stream) {
  const float* x   = (const float*)d_in[0];
  const float* h0  = (const float*)d_in[1];
  const float* c0  = (const float*)d_in[2];
  const float* wih = (const float*)d_in[3];
  const float* whh = (const float*)d_in[4];
  const float* bih = (const float*)d_in[5];
  const float* bhh = (const float*)d_in[6];
  float* out = (float*)d_out;

  // ---- workspace carve-up (~621 MB) ----
  char* ws = (char*)d_ws;
  size_t off = 0;
  float*  xg     = (float*)(ws + off);  off += (size_t)T_STEPS * BATCH * GSZ * 4;
  bf16_t* x_bf   = (bf16_t*)(ws + off); off += (size_t)T_STEPS * BATCH * ISZ * 2;
  bf16_t* wih_bf = (bf16_t*)(ws + off); off += (size_t)GSZ * ISZ * 2;
  bf16_t* whh_bf = (bf16_t*)(ws + off); off += (size_t)GSZ * HSZ * 2;
  bf16_t* hbuf   = (bf16_t*)(ws + off); off += (size_t)2 * BATCH * HSZ * 2;
  float*  cstate = (float*)(ws + off);  off += (size_t)BATCH * HSZ * 4;

  // ---- fp32 -> bf16 conversions ----
  {
    int n4 = T_STEPS * BATCH * ISZ / 4;
    cvt4_f32_bf16<<<(n4 + 255) / 256, 256, 0, stream>>>(
        (const float4*)x, (bf16x4*)x_bf, n4);
    n4 = GSZ * ISZ / 4;
    cvt4_f32_bf16<<<(n4 + 255) / 256, 256, 0, stream>>>(
        (const float4*)wih, (bf16x4*)wih_bf, n4);
    n4 = GSZ * HSZ / 4;
    cvt4_f32_bf16<<<(n4 + 255) / 256, 256, 0, stream>>>(
        (const float4*)whh, (bf16x4*)whh_bf, n4);
  }
  init_state<<<(BATCH * HSZ + 255) / 256, 256, 0, stream>>>(
      h0, c0, hbuf, cstate, BATCH * HSZ);

  // ---- phase 1: big GEMM ----
  gemm_xg<<<65536, 256, 0, stream>>>(x_bf, wih_bf, bih, bhh, xg);

  // ---- phase 2: sequential scan ----
  const size_t BH  = (size_t)BATCH * HSZ;      // 65536
  const size_t TBH = (size_t)T_STEPS * BH;     // 33554432
  float* o_out = out;                          // outputs [T,B,H]
  float* hT    = out + TBH;                    // h_t [B,H]
  float* cT    = out + TBH + BH;               // c_t [B,H]
  float* oi    = out + TBH + 2 * BH;           // i_g [T,B,H]
  float* of    = oi + TBH;
  float* og    = of + TBH;
  float* oo    = og + TBH;

  for (int t = 0; t < T_STEPS; ++t) {
    const float*  xg_t = xg + (size_t)t * BATCH * GSZ;
    const bf16_t* hin  = hbuf + (size_t)(t & 1) * BH;
    bf16_t*       hout = hbuf + (size_t)((t + 1) & 1) * BH;
    bool last = (t == T_STEPS - 1);
    lstm_step<<<32, 256, 0, stream>>>(
        xg_t, hin, whh_bf, hout, cstate,
        o_out + (size_t)t * BH, oi + (size_t)t * BH, of + (size_t)t * BH,
        og + (size_t)t * BH, oo + (size_t)t * BH,
        last ? hT : (float*)nullptr, last ? cT : (float*)nullptr);
  }
}
